// TemporalAttention_27676769255590
// MI455X (gfx1250) — compile-verified, matrix operand order checked by reading, not run on hardware
//
#include <hip/hip_runtime.h>

// ---------- CDNA5 WMMA types ----------
typedef _Float16 v16h __attribute__((ext_vector_type(16)));
typedef _Float16 v8h  __attribute__((ext_vector_type(8)));
typedef float    v8f  __attribute__((ext_vector_type(8)));

// Problem constants (match reference setup_inputs)
#define BB 8
#define SS 2048
#define DD 1024

// ---------------------------------------------------------------------------
// Block-GEMM geometry: 256 threads = 8 waves, wave grid 4(M) x 2(N).
// Block tile: 128(M) x 128(N), K staged in chunks of 64.
// LDS tile: 128 rows x 64 halves, row padded to 72 halves (144 B = 36 DWORDs)
// -> b128 fragment reads are LDS bank-conflict-free (36*l mod 64 all distinct).
// 4 tiles (A,B double-buffered) = 72 KB LDS (gfx1250 WGP has 320 KB).
// ---------------------------------------------------------------------------
#define T_ROWS 128
#define T_KH   64
#define T_LDH  72                   // halves per padded LDS row
#define T_H    (T_ROWS * T_LDH)     // halves per tile (9216)

__device__ __forceinline__ unsigned lds_off(const void* p) {
    return (unsigned)(unsigned long long)p;   // low 32 bits of AS(3) addr = LDS byte offset
}

// CDNA5 async copy: 16 B per lane, global -> LDS, tracked by ASYNCcnt.
__device__ __forceinline__ void async_ld_b128(unsigned lds, const _Float16* g) {
    asm volatile("global_load_async_to_lds_b128 %0, %1, off"
                 :: "v"(lds), "v"(g) : "memory");
}
#define WAIT_ASYNC_8 asm volatile("s_wait_asynccnt 8" ::: "memory")
#define WAIT_ASYNC_0 asm volatile("s_wait_asynccnt 0" ::: "memory")

// ---------------------------------------------------------------------------
// A-style 16x32 f16 fragment from row-major [16 x ld] memory (global or LDS).
// Lane layout (ISA 7.12.2): lanes 0-15 row=lane, K 0..7 & 16..23;
// lanes 16-31 row=lane-16, K 8..15 & 24..31  -> two contiguous 16-B loads.
// WMMA B operand (32x16, lane = output column) has the identical per-lane
// layout, so B fragments load with this helper from the operand's transpose.
// ---------------------------------------------------------------------------
__device__ __forceinline__ v16h load_frag(const _Float16* base, int ld) {
    const int lane = threadIdx.x & 31;
    const int row  = lane & 15;
    const int koff = (lane >> 4) << 3;            // 0 or 8
    const _Float16* p = base + (long)row * ld + koff;
    v8h lo = *reinterpret_cast<const v8h*>(p);
    v8h hi = *reinterpret_cast<const v8h*>(p + 16);
    return __builtin_shufflevector(lo, hi, 0,1,2,3,4,5,6,7,8,9,10,11,12,13,14,15);
}

__device__ __forceinline__ v8f wmma_f16(v16h a, v16h b, v8f c) {
    return __builtin_amdgcn_wmma_f32_16x16x32_f16(false, a, false, b, (short)0, c, false, false);
}

// ---------------------------------------------------------------------------
// Core: C(128x128) += A(128xK) * Bt(128xK)^T with LDS double buffering.
// Ab: block rows of A (lda), Bb: block rows of Bt (ldb) = output columns.
// c[i][j]: wave tile — rows mw*32+i*16, cols nw*64+j*16 within the block.
// ---------------------------------------------------------------------------
__device__ __forceinline__ void gemm_tiles(const _Float16* __restrict__ Ab, int lda,
                                           const _Float16* __restrict__ Bb, int ldb,
                                           int K, _Float16* sm, v8f c[2][4]) {
    const int tid = threadIdx.x;
    const int w = tid >> 5, mw = w & 3, nw = w >> 2;
    const int nst = K / T_KH;

    auto issue = [&](int buf, int kb) {
        _Float16* sA = sm + buf * 2 * T_H;
        _Float16* sB = sA + T_H;
        const unsigned aL = lds_off(sA), bL = lds_off(sB);
        #pragma unroll
        for (int i = 0; i < 4; ++i) {                 // 1024 tasks / 256 threads
            const int task = tid + i * 256;
            const int r = task >> 3, ch = task & 7;   // row, 16-B chunk (8 per 128-B row)
            async_ld_b128(aL + r * (T_LDH * 2) + ch * 16,
                          Ab + (long)r * lda + kb + ch * 8);
        }
        #pragma unroll
        for (int i = 0; i < 4; ++i) {
            const int task = tid + i * 256;
            const int r = task >> 3, ch = task & 7;
            async_ld_b128(bL + r * (T_LDH * 2) + ch * 16,
                          Bb + (long)r * ldb + kb + ch * 8);
        }
    };

    issue(0, 0);
    for (int s = 0; s < nst; ++s) {
        if (s + 1 < nst) {                 // prefetch next stage, then drain current
            issue((s + 1) & 1, (s + 1) * T_KH);
            WAIT_ASYNC_8;                  // async loads complete in order
        } else {
            WAIT_ASYNC_0;
        }
        __syncthreads();
        const _Float16* sA = sm + (s & 1) * 2 * T_H;
        const _Float16* sB = sA + T_H;
        #pragma unroll
        for (int kc = 0; kc < 2; ++kc) {   // two K=32 sub-chunks per stage
            v16h a0 = load_frag(sA + (mw * 32 +  0) * T_LDH + kc * 32, T_LDH);
            v16h a1 = load_frag(sA + (mw * 32 + 16) * T_LDH + kc * 32, T_LDH);
            #pragma unroll
            for (int j = 0; j < 4; ++j) {
                v16h bf = load_frag(sB + (nw * 64 + j * 16) * T_LDH + kc * 32, T_LDH);
                c[0][j] = wmma_f16(a0, bf, c[0][j]);
                c[1][j] = wmma_f16(a1, bf, c[1][j]);
            }
        }
        __syncthreads();                   // everyone done reading before overwrite
    }
}

// ---------------------------------------------------------------------------
// Kernel 1: f32 -> f16 elementwise convert (x -> x16)
// ---------------------------------------------------------------------------
__global__ void k_cvt(const float* __restrict__ in, _Float16* __restrict__ out, long n) {
    long i = (long)blockIdx.x * blockDim.x + threadIdx.x;
    long stride = (long)gridDim.x * blockDim.x;
    for (; i < n; i += stride) out[i] = (_Float16)in[i];
}

// ---------------------------------------------------------------------------
// Kernel 2: tiled transpose + convert: f32 [R][C] -> f16 [C][R], batched.
// ---------------------------------------------------------------------------
__global__ void k_tconv(const float* __restrict__ in, _Float16* __restrict__ out,
                        int R, int C, long inBatch, long outBatch) {
    __shared__ float tile[32][33];
    const int b  = blockIdx.z;
    const int c0 = blockIdx.x * 32;
    const int r0 = blockIdx.y * 32;
    const int tx = threadIdx.x, ty = threadIdx.y;       // block (32,8)
    const float* ib = in + (long)b * inBatch;
    _Float16*    ob = out + (long)b * outBatch;
    #pragma unroll
    for (int i = ty; i < 32; i += 8)
        tile[i][tx] = ib[(long)(r0 + i) * C + (c0 + tx)];
    __syncthreads();
    #pragma unroll
    for (int i = ty; i < 32; i += 8)
        ob[(long)(c0 + i) * R + (r0 + tx)] = (_Float16)tile[tx][i];
}

// ---------------------------------------------------------------------------
// Kernel 3: Q GEMM.  q16[M][N] = x16[M][K] * W, Bt = W^T f16 [N][K].
// ---------------------------------------------------------------------------
__global__ void __launch_bounds__(256)
k_gemm_q(const _Float16* __restrict__ A, const _Float16* __restrict__ Bt,
         _Float16* __restrict__ Dst, int M, int N, int K) {
    __shared__ _Float16 sm[4 * T_H];
    const int bn = blockIdx.x & 7, bm = blockIdx.x >> 3;     // N/128 = 8 strips
    const _Float16* Ab = A  + (long)bm * 128 * K;
    const _Float16* Bb = Bt + (long)bn * 128 * K;

    v8f c[2][4] = {};
    gemm_tiles(Ab, K, Bb, K, K, sm, c);

    const int lane = threadIdx.x & 31;
    const int w = threadIdx.x >> 5, mw = w & 3, nw = w >> 2;
    const int cr0 = (lane >> 4) << 3;
    const int col = bn * 128 + nw * 64 + (lane & 15);
    #pragma unroll
    for (int i = 0; i < 2; ++i) {
        const int r0 = bm * 128 + mw * 32 + i * 16 + cr0;
        #pragma unroll
        for (int j = 0; j < 4; ++j)
            #pragma unroll
            for (int e = 0; e < 8; ++e)
                Dst[(long)(r0 + e) * N + col + j * 16] = (_Float16)c[i][j][e];
    }
}

// ---------------------------------------------------------------------------
// Kernel 4: scores.  S[b][q][k] = (1/32) * q16[b][q][:] . x16[b][k][:]
// ---------------------------------------------------------------------------
__global__ void __launch_bounds__(256)
k_scores(const _Float16* __restrict__ Q, const _Float16* __restrict__ Xk,
         _Float16* __restrict__ S, int Sq, int Sk, int D, float scale) {
    __shared__ _Float16 sm[4 * T_H];
    const int b  = blockIdx.y;
    const int bm = blockIdx.x >> 4, bn = blockIdx.x & 15;    // 16x16 blocks of 128
    const _Float16* Ab = Q  + (long)b * Sq * D + (long)bm * 128 * D;
    const _Float16* Bb = Xk + (long)b * Sk * D + (long)bn * 128 * D;
    _Float16*       s0 = S  + (long)b * Sq * Sk;

    v8f c[2][4] = {};
    gemm_tiles(Ab, D, Bb, D, D, sm, c);

    const int lane = threadIdx.x & 31;
    const int w = threadIdx.x >> 5, mw = w & 3, nw = w >> 2;
    const int cr0 = (lane >> 4) << 3;
    const int col = bn * 128 + nw * 64 + (lane & 15);
    #pragma unroll
    for (int i = 0; i < 2; ++i) {
        const int r0 = bm * 128 + mw * 32 + i * 16 + cr0;
        #pragma unroll
        for (int j = 0; j < 4; ++j)
            #pragma unroll
            for (int e = 0; e < 8; ++e)
                s0[(long)(r0 + e) * Sk + col + j * 16] = (_Float16)(c[i][j][e] * scale);
    }
}

// ---------------------------------------------------------------------------
// Kernel 5: row softmax in place on f16 scores; one wave32 per row.
// ---------------------------------------------------------------------------
__global__ void k_softmax(_Float16* __restrict__ P, int Sk) {
    const int row  = (int)((blockIdx.x * blockDim.x + threadIdx.x) >> 5);
    const int lane = threadIdx.x & 31;
    _Float16* r = P + (long)row * Sk;
    const int nper = Sk / 32;                   // 64

    float buf[64];
    float mx = -3.0e30f;
    #pragma unroll
    for (int i = 0; i < nper; ++i) {
        buf[i] = (float)r[lane + i * 32];
        mx = fmaxf(mx, buf[i]);
    }
    #pragma unroll
    for (int off = 16; off > 0; off >>= 1)
        mx = fmaxf(mx, __shfl_xor(mx, off, 32));

    float sum = 0.f;
    #pragma unroll
    for (int i = 0; i < nper; ++i) {
        buf[i] = __expf(buf[i] - mx);
        sum += buf[i];
    }
    #pragma unroll
    for (int off = 16; off > 0; off >>= 1)
        sum += __shfl_xor(sum, off, 32);

    const float inv = 1.0f / sum;
    #pragma unroll
    for (int i = 0; i < nper; ++i)
        r[lane + i * 32] = (_Float16)(buf[i] * inv);
}

// ---------------------------------------------------------------------------
// Kernel 6: out = P * V + bias.  A = P rows, B = xT16 rows (= output columns).
// ---------------------------------------------------------------------------
__global__ void __launch_bounds__(256)
k_out(const _Float16* __restrict__ P, const _Float16* __restrict__ Vt,
      const float* __restrict__ bias, float* __restrict__ out,
      int Sq, int Sk, int D) {
    __shared__ _Float16 sm[4 * T_H];
    const int b  = blockIdx.y;
    const int bm = blockIdx.x >> 3, bn = blockIdx.x & 7;     // 16 x 8 blocks of 128
    const _Float16* Ab = P  + (long)b * Sq * Sk + (long)bm * 128 * Sk;
    const _Float16* Bb = Vt + (long)b * D * Sk + (long)bn * 128 * Sk;
    float*          o0 = out + (long)b * Sq * D;

    v8f c[2][4] = {};
    gemm_tiles(Ab, Sk, Bb, Sk, Sk, sm, c);

    const int lane = threadIdx.x & 31;
    const int w = threadIdx.x >> 5, mw = w & 3, nw = w >> 2;
    const int cr0 = (lane >> 4) << 3;
    const int col = bn * 128 + nw * 64 + (lane & 15);
    #pragma unroll
    for (int i = 0; i < 2; ++i) {
        const int r0 = bm * 128 + mw * 32 + i * 16 + cr0;
        #pragma unroll
        for (int j = 0; j < 4; ++j) {
            const float bv = bias[col + j * 16];
            #pragma unroll
            for (int e = 0; e < 8; ++e)
                o0[(long)(r0 + e) * D + col + j * 16] = c[i][j][e] + bv;
        }
    }
}

// ---------------------------------------------------------------------------
extern "C" void kernel_launch(void* const* d_in, const int* in_sizes, int n_in,
                              void* d_out, int out_size, void* d_ws, size_t ws_size,
                              hipStream_t stream) {
    const float* x    = (const float*)d_in[0];   // [B,S,D]
    const float* W    = (const float*)d_in[1];   // [D,D]
    const float* bias = (const float*)d_in[2];   // [D]
    float* out = (float*)d_out;

    // workspace layout (f16), all regions fully written before read
    char* ws = (char*)d_ws;
    _Float16* x16  = (_Float16*)(ws);                                 // B*S*D
    _Float16* xT16 = (_Float16*)(ws + 33554432L);                     // B*D*S
    _Float16* q16  = (_Float16*)(ws + 67108864L);                     // B*S*D
    _Float16* wT16 = (_Float16*)(ws + 100663296L);                    // D*D
    _Float16* P16  = (_Float16*)(ws + 102760448L);                    // B*S*S

    const long nX = (long)BB * SS * DD;

    // 1) x -> f16
    k_cvt<<<65536, 256, 0, stream>>>(x, x16, nX);

    // 2) x^T per batch (V operand), W^T (Q GEMM B operand)
    {
        dim3 blk(32, 8);
        dim3 grdX(DD / 32, SS / 32, BB);
        k_tconv<<<grdX, blk, 0, stream>>>(x, xT16, SS, DD, (long)SS * DD, (long)DD * SS);
        dim3 grdW(DD / 32, DD / 32, 1);
        k_tconv<<<grdW, blk, 0, stream>>>(W, wT16, DD, DD, 0L, 0L);
    }

    // 3) Q = x @ W   (M = B*S = 16384, N = K = D): 128x8 blocks of 128x128
    k_gemm_q<<<(BB * SS / 128) * (DD / 128), 256, 0, stream>>>(
        x16, wT16, q16, BB * SS, DD, DD);

    // 4) scores = (Q . x^T) / sqrt(D): per batch 16x16 blocks of 128x128
    k_scores<<<dim3((SS / 128) * (SS / 128), BB), 256, 0, stream>>>(
        q16, x16, P16, SS, SS, DD, 1.0f / 32.0f);

    // 5) softmax rows (B*S rows, one wave each)
    k_softmax<<<(BB * SS) / 8, 256, 0, stream>>>(P16, SS);

    // 6) out = P @ x + bias: per batch 16x8 blocks of 128x128
    k_out<<<dim3((SS / 128) * (DD / 128), BB), 256, 0, stream>>>(
        P16, xT16, bias, out, SS, SS, DD);
}
